// SAN_44796508897835
// MI455X (gfx1250) — matrix operand order repaired and checked by reading, not compiled, not run on hardware
//
#include <hip/hip_runtime.h>
#include <math.h>

// ---------------------------------------------------------------------------
// CDNA5 (gfx1250, wave32) types for v_wmma_f32_16x16x32_bf16
// ---------------------------------------------------------------------------
typedef __attribute__((ext_vector_type(16))) __bf16         v16bf;
typedef __attribute__((ext_vector_type(8)))  float          v8f;
typedef __attribute__((ext_vector_type(16))) unsigned short v16us;
typedef __attribute__((ext_vector_type(8)))  unsigned short v8us;

__device__ __forceinline__ unsigned short f2bf(float f) {
    unsigned int u = __float_as_uint(f);
    u += 0x7FFFu + ((u >> 16) & 1u);          // round-to-nearest-even
    return (unsigned short)(u >> 16);
}
__device__ __forceinline__ float bf2f(unsigned short u) {
    return __uint_as_float((unsigned int)u << 16);
}

// ---------------------------------------------------------------------------
// Tiled GEMM, bf16 operands, fp32 accum, double-buffered LDS with CDNA5
// async global->LDS prefetch (ASYNCcnt) for the A tile:
//   acc = op(A)(MxK) @ B(Kx128);  alpha = alphaC * (alphaPtr ? *alphaPtr : 1)
//   mode 0: out = alpha*acc
//   mode 1: out = Cadd + alpha*acc            (harmonic step y - eps*L@y)
//   mode 2: out = Cout + alpha*acc            (term accumulation)
//   mode 3: out = relu(Cout + alpha*acc)      (final fused term)
//   Cout (f32) nullable (bf16-only result), CoutBf (bf16 mirror) nullable.
// M % 128 == 0, K % 32 == 0 for all calls here.
// ---------------------------------------------------------------------------
#define BLK_M 128
#define BLK_K 32

__global__ __launch_bounds__(256, 1)
void gemm128_bf16(const unsigned short* __restrict__ A, int lda, int transA,
                  const unsigned short* __restrict__ B,
                  const float* __restrict__ Cadd,
                  float* __restrict__ Cout,
                  unsigned short* __restrict__ CoutBf,
                  int M, int K,
                  float alphaC, const float* __restrict__ alphaPtr,
                  int mode)
{
    // ping-pong tiles: A 128x32 (80B row stride, 16B aligned), B^T 128x32 (96B stride)
    __shared__ alignas(128) unsigned short As[2][BLK_M][BLK_K + 8];
    __shared__ alignas(128) unsigned short Bs[2][128][BLK_K + 16];

    const int tid  = threadIdx.x;
    const int lane = tid & 31;
    const int wv   = tid >> 5;                 // 8 waves -> 8 x 16 rows
    const int row0 = blockIdx.x * BLK_M;

    float alpha = alphaC;
    if (alphaPtr) alpha *= alphaPtr[0];

    v8f acc[8];
    const v8f vz = {0.f,0.f,0.f,0.f,0.f,0.f,0.f,0.f};
#pragma unroll
    for (int i = 0; i < 8; ++i) acc[i] = vz;

    // stage tile starting at column k0 into LDS buffer `buf`
    auto stage = [&](int k0, int buf) {
        if (!transA) {
            // rows are contiguous bf16: 4 x 16B chunks per row, 512 chunks total,
            // moved by the async global->LDS engine (no VGPR data path).
#pragma unroll
            for (int t = 0; t < 2; ++t) {
                int i  = tid + t * 256;            // chunk id 0..511
                int r  = i >> 2;                   // tile row
                int ch = (i & 3) * 8;              // chunk offset in ushorts
                unsigned long long ga = (unsigned long long)(size_t)A
                    + 2ull * ((unsigned long long)(unsigned)(row0 + r) * (unsigned long long)(unsigned)lda
                              + (unsigned long long)(unsigned)k0)
                    + 2ull * (unsigned long long)ch;
                unsigned ldsa = (unsigned)(size_t)&As[buf][r][ch];
                asm volatile("global_load_async_to_lds_b128 %0, %1, off"
                             :: "v"(ldsa), "v"(ga) : "memory");
            }
        } else {
            for (int i = tid; i < BLK_M * BLK_K; i += 256) {
                int c = i >> 7, r = i & 127;       // coalesced along M
                As[buf][r][c] = A[(size_t)(k0 + c) * lda + (row0 + r)];
            }
        }
        // B tile (32 x 128 bf16) transposed into LDS: Bs[col][k]
        const unsigned int* Bu = (const unsigned int*)B;   // 2 cols per dword
        for (int i = tid; i < 32 * 64; i += 256) {
            int r = i >> 6, cp = i & 63;
            unsigned int w = Bu[(size_t)(k0 + r) * 64 + cp];
            Bs[buf][cp * 2][r]     = (unsigned short)(w & 0xFFFFu);
            Bs[buf][cp * 2 + 1][r] = (unsigned short)(w >> 16);
        }
    };

    const int ntiles = K >> 5;

    stage(0, 0);
    if (!transA) asm volatile("s_wait_asynccnt 0x0" ::: "memory");
    __syncthreads();

    for (int t = 0; t < ntiles; ++t) {
        const int cur = t & 1;
        if (t + 1 < ntiles) stage((t + 1) << 5, cur ^ 1);   // prefetch next tile

        // ---- A fragment (contiguous k = 16*(lane/16) + s, same order as B) ----
        const int arow  = wv * 16 + (lane & 15);
        const int abase = (lane >> 4) * 16;    // lanes 0-15: K 0..15, 16-31: K 16..31
        v8us alo = *(const v8us*)&As[cur][arow][abase];
        v8us ahi = *(const v8us*)&As[cur][arow][abase + 8];
        v16us auv;
#pragma unroll
        for (int e = 0; e < 8; ++e) { auv[e] = alo[e]; auv[8 + e] = ahi[e]; }
        v16bf afrag = __builtin_bit_cast(v16bf, auv);

        // ---- preload all 8 B fragments, then 8 back-to-back WMMAs ----
        const int kb = (lane >> 4) * 16;       // lanes 0-15: K 0..15, 16-31: K 16..31
        v16us buv[8];
#pragma unroll
        for (int nb = 0; nb < 8; ++nb)
            buv[nb] = *(const v16us*)&Bs[cur][nb * 16 + (lane & 15)][kb];
#pragma unroll
        for (int nb = 0; nb < 8; ++nb) {
            v16bf bfrag = __builtin_bit_cast(v16bf, buv[nb]);
            acc[nb] = __builtin_amdgcn_wmma_f32_16x16x32_bf16(
                false, afrag, false, bfrag, (short)0, acc[nb], false, false);
        }

        if (!transA && t + 1 < ntiles)
            asm volatile("s_wait_asynccnt 0x0" ::: "memory");
        __syncthreads();
    }

    // ---- epilogue (C/D layout: col = lane&15, rows (lane>>4)*8 + r) ----
    const int crow0 = row0 + wv * 16 + (lane >> 4) * 8;
    const int ccol  = lane & 15;
#pragma unroll
    for (int nb = 0; nb < 8; ++nb) {
        int col = nb * 16 + ccol;
#pragma unroll
        for (int r = 0; r < 8; ++r) {
            int row = crow0 + r;
            if (row < M) {
                size_t o = (size_t)row * 128 + col;
                float v = alpha * acc[nb][r];
                float outv;
                if (mode == 1)      outv = Cadd[o] + v;
                else if (mode == 2) outv = Cout[o] + v;
                else if (mode == 3) outv = fmaxf(Cout[o] + v, 0.f);
                else                outv = v;
                if (Cout)   Cout[o]   = outv;
                if (CoutBf) CoutBf[o] = f2bf(outv);
            }
        }
    }
}

// ---------------------------------------------------------------------------
// Helper kernels
// ---------------------------------------------------------------------------
__global__ void cvt_bf16(const float* __restrict__ in, unsigned short* __restrict__ out, size_t n) {
    size_t i = (size_t)blockIdx.x * blockDim.x + threadIdx.x;
    size_t stride = (size_t)gridDim.x * blockDim.x;
    for (; i < n; i += stride) out[i] = f2bf(in[i]);
}

__global__ void wsum3_bf16(const float* __restrict__ W, unsigned short* __restrict__ out) {
    int i = blockIdx.x * blockDim.x + threadIdx.x;       // 16384 elements
    if (i < 16384) out[i] = f2bf(W[i] + W[16384 + i] + W[32768 + i]);
}

__global__ void fill_val(float* __restrict__ p, int n, float v) {
    int i = blockIdx.x * blockDim.x + threadIdx.x;
    if (i < n) p[i] = v;
}

// u[k] = z[k]·w[0:128], v[k] = z[k]·w[128:256]   (one wave per row)
__global__ void attn_uv(const float* __restrict__ z, int n,
                        const float* __restrict__ w,
                        float* __restrict__ u, float* __restrict__ v)
{
    int wv = threadIdx.x >> 5, lane = threadIdx.x & 31;
    int row = blockIdx.x * 4 + wv;                       // blockDim = 128
    if (row >= n) return;
    float su = 0.f, sv = 0.f;
    for (int c = lane; c < 128; c += 32) {
        float x = z[(size_t)row * 128 + c];
        su += x * w[c];
        sv += x * w[128 + c];
    }
    for (int off = 16; off > 0; off >>= 1) {
        su += __shfl_down(su, off, 32);
        sv += __shfl_down(sv, off, 32);
    }
    if (lane == 0) { u[row] = su; v[row] = sv; }
}

// A[i,j] = nonzero(Lpat[i,j]) ? sigmoid(u[j] + v[i] + b) : 0   (bf16 out)
__global__ void attn_mat(const float* __restrict__ Lpat, int n,
                         const float* __restrict__ u, const float* __restrict__ v,
                         const float* __restrict__ bptr, unsigned short* __restrict__ Aout)
{
    int i = blockIdx.y;
    int j = blockIdx.x * blockDim.x + threadIdx.x;
    if (j >= n) return;
    size_t idx = (size_t)i * n + j;
    float s = 0.f;
    if (Lpat[idx] != 0.f) {
        float t = u[j] + v[i] + bptr[0];
        s = 1.f / (1.f + __expf(-t));
    }
    Aout[idx] = f2bf(s);
}

// bigb = bf16(L1d + L1u)
__global__ void addmat_bf16(const float* __restrict__ a, const float* __restrict__ b,
                            unsigned short* __restrict__ c, size_t n)
{
    size_t i = (size_t)blockIdx.x * blockDim.x + threadIdx.x;
    size_t stride = (size_t)gridDim.x * blockDim.x;
    for (; i < n; i += stride) c[i] = f2bf(a[i] + b[i]);
}

// Gershgorin: outmax = max_i sum_j |L[i,j]|   (bf16 matrix; atomicMax on bits)
__global__ void row_abs_max(const unsigned short* __restrict__ L, int n, float* __restrict__ outmax) {
    int wv = threadIdx.x >> 5, lane = threadIdx.x & 31;
    int row = blockIdx.x * 8 + wv;                       // blockDim = 256
    if (row >= n) return;
    float s = 0.f;
    for (int j = lane; j < n; j += 32) s += fabsf(bf2f(L[(size_t)row * n + j]));
    for (int off = 16; off > 0; off >>= 1) s += __shfl_down(s, off, 32);
    if (lane == 0) atomicMax((int*)outmax, __float_as_int(s));
}

// y = (L + sigma*I) x    (bf16 matrix, f32 vectors)
__global__ void symmv_shift(const unsigned short* __restrict__ L, int n,
                            const float* __restrict__ sig,
                            const float* __restrict__ x, float* __restrict__ y)
{
    int wv = threadIdx.x >> 5, lane = threadIdx.x & 31;
    int row = blockIdx.x * 8 + wv;                       // blockDim = 256
    if (row >= n) return;
    float s = 0.f;
    for (int j = lane; j < n; j += 32) s += bf2f(L[(size_t)row * n + j]) * x[j];
    for (int off = 16; off > 0; off >>= 1) s += __shfl_down(s, off, 32);
    if (lane == 0) y[row] = s + sig[0] * x[row];
}

// deterministic single-block sum of squares
__global__ void reduce_ssq(const float* __restrict__ y, int n, float* __restrict__ out) {
    __shared__ float sm[256];
    float s = 0.f;
    for (int i = threadIdx.x; i < n; i += 256) { float t = y[i]; s += t * t; }
    sm[threadIdx.x] = s; __syncthreads();
    for (int off = 128; off > 0; off >>= 1) {
        if ((int)threadIdx.x < off) sm[threadIdx.x] += sm[threadIdx.x + off];
        __syncthreads();
    }
    if (threadIdx.x == 0) out[0] = sm[0];
}

__global__ void normalize_vec(const float* __restrict__ y, int n,
                              const float* __restrict__ ss, float* __restrict__ x)
{
    int i = blockIdx.x * blockDim.x + threadIdx.x;
    if (i < n) x[i] = y[i] * rsqrtf(fmaxf(ss[0], 1e-30f));
}

__global__ void finalize_eps(float* __restrict__ scal) {
    if (threadIdx.x == 0 && blockIdx.x == 0) {
        float lmax = sqrtf(fmaxf(scal[1], 0.f)) - scal[0];
        scal[2] = (lmax > 0.f) ? (1.f / lmax) : 0.1f;
    }
}

// ---------------------------------------------------------------------------
// Host orchestration
// ---------------------------------------------------------------------------
extern "C" void kernel_launch(void* const* d_in, const int* in_sizes, int n_in,
                              void* d_out, int out_size, void* d_ws, size_t ws_size,
                              hipStream_t stream)
{
    const int N0 = 1024, N1 = 3072, N2 = 2048, C = 128;
    const float* z0   = (const float*)d_in[0];
    const float* z1   = (const float*)d_in[1];
    const float* z2   = (const float*)d_in[2];
    const float* L0   = (const float*)d_in[3];
    const float* L1d  = (const float*)d_in[4];
    const float* L1u  = (const float*)d_in[5];
    const float* L2   = (const float*)d_in[6];
    const float* B1   = (const float*)d_in[7];
    const float* B2   = (const float*)d_in[8];
    const float* W_d  = (const float*)d_in[9];
    const float* W_u  = (const float*)d_in[10];
    const float* W_h  = (const float*)d_in[11];
    const float* W_b1 = (const float*)d_in[12];
    const float* W_b2 = (const float*)d_in[13];
    const float* adw  = (const float*)d_in[14];
    const float* adb  = (const float*)d_in[15];
    const float* auw  = (const float*)d_in[16];
    const float* aub  = (const float*)d_in[17];
    (void)in_sizes; (void)n_in; (void)out_size; (void)ws_size;

    float* out0 = (float*)d_out;
    float* out1 = out0 + (size_t)N0 * C;
    float* out2 = out1 + (size_t)N1 * C;

    // ---- workspace carve-up (byte based, 256B aligned blocks) ----
    char* base = (char*)d_ws;
    size_t off = 0;
    auto allocF = [&](size_t n) { float* p = (float*)(base + off);
                                  off += ((n * 4 + 255) & ~(size_t)255); return p; };
    auto allocU = [&](size_t n) { unsigned short* p = (unsigned short*)(base + off);
                                  off += ((n * 2 + 255) & ~(size_t)255); return p; };
    // f32 buffers
    float* uvec = allocF(N1);
    float* vvec = allocF(N1);
    float* scal = allocF(64);                 // [0]=sigma [1]=ssq [2]=eps
    float* pvx  = allocF(N1);
    float* pvy  = allocF(N1);
    float* yA_f = allocF((size_t)N1 * C);
    float* yB_f = allocF((size_t)N1 * C);
    // bf16 buffers
    unsigned short* zb0  = allocU((size_t)N0 * C);
    unsigned short* zb1  = allocU((size_t)N1 * C);
    unsigned short* zb2  = allocU((size_t)N2 * C);
    unsigned short* Lb0  = allocU((size_t)N0 * N0);
    unsigned short* Lb2  = allocU((size_t)N2 * N2);
    unsigned short* B1b  = allocU((size_t)N0 * N1);
    unsigned short* B2b  = allocU((size_t)N1 * N2);
    unsigned short* Whb  = allocU(16384);
    unsigned short* Wdb  = allocU(16384);
    unsigned short* Wub  = allocU(16384);
    unsigned short* Wb1b = allocU(16384);
    unsigned short* Wb2b = allocU(16384);
    unsigned short* yA_b = allocU((size_t)N1 * C);
    unsigned short* yB_b = allocU((size_t)N1 * C);
    unsigned short* T_b  = allocU((size_t)N1 * C);
    unsigned short* bigb = allocU((size_t)N1 * N1);   // A_d -> A_u -> L1d+L1u

    auto gemm = [&](const unsigned short* A, int lda, int tA, const unsigned short* B_,
                    const float* Cadd, float* Cout, unsigned short* CoutBf,
                    int M, int K, float aC, const float* aP, int mode) {
        gemm128_bf16<<<dim3(M / 128), dim3(256), 0, stream>>>(
            A, lda, tA, B_, Cadd, Cout, CoutBf, M, K, aC, aP, mode);
    };
    auto cvt = [&](const float* in, unsigned short* out, size_t n) {
        cvt_bf16<<<dim3(2048), dim3(256), 0, stream>>>(in, out, n);
    };

    // shifted power iteration on bf16 matrix -> scal[2] = eps = 1/lambda_max
    auto power_eps = [&](const unsigned short* Lb, int n) {
        fill_val<<<dim3(1), dim3(64), 0, stream>>>(scal, 8, 0.f);
        row_abs_max<<<dim3(n / 8), dim3(256), 0, stream>>>(Lb, n, scal);
        fill_val<<<dim3((n + 255) / 256), dim3(256), 0, stream>>>(pvx, n, 1.f);
        for (int it = 0; it < 40; ++it) {
            symmv_shift<<<dim3(n / 8), dim3(256), 0, stream>>>(Lb, n, scal, pvx, pvy);
            reduce_ssq<<<dim3(1), dim3(256), 0, stream>>>(pvy, n, scal + 1);
            normalize_vec<<<dim3((n + 255) / 256), dim3(256), 0, stream>>>(pvy, n, scal + 1, pvx);
        }
        finalize_eps<<<dim3(1), dim3(64), 0, stream>>>(scal);
    };

    // y <- (I - eps*L)^16 z  via 16 fused GEMM steps; returns final bf16 buffer
    auto harmonic = [&](const unsigned short* Lb, int n,
                        const float* z_f, const unsigned short* z_b) -> const unsigned short* {
        const float* sf = z_f; const unsigned short* sb = z_b;
        float* df = yA_f; unsigned short* db = yA_b;
        for (int t = 0; t < 16; ++t) {
            gemm(Lb, n, 0, sb, sf, df, db, n, n, -1.f, scal + 2, 1);
            sf = df; sb = db;
            if (df == yA_f) { df = yB_f; db = yB_b; }
            else            { df = yA_f; db = yA_b; }
        }
        return sb;
    };

    // ---- one-time bf16 conversions / collapsed weights ----
    cvt(z0, zb0, (size_t)N0 * C);
    cvt(z1, zb1, (size_t)N1 * C);
    cvt(z2, zb2, (size_t)N2 * C);
    cvt(L0, Lb0, (size_t)N0 * N0);
    cvt(L2, Lb2, (size_t)N2 * N2);
    cvt(B1, B1b, (size_t)N0 * N1);
    cvt(B2, B2b, (size_t)N1 * N2);
    cvt(W_h, Whb, 16384);
    wsum3_bf16<<<dim3(64), dim3(256), 0, stream>>>(W_d,  Wdb);
    wsum3_bf16<<<dim3(64), dim3(256), 0, stream>>>(W_u,  Wub);
    wsum3_bf16<<<dim3(64), dim3(256), 0, stream>>>(W_b1, Wb1b);
    wsum3_bf16<<<dim3(64), dim3(256), 0, stream>>>(W_b2, Wb2b);

    // ================= z0_out =================
    power_eps(Lb0, N0);
    const unsigned short* yf0 = harmonic(Lb0, N0, z0, zb0);
    gemm(yf0, 128, 0, Whb, nullptr, out0, nullptr, N0, 128, 1.f, nullptr, 0);     // init
    gemm(Lb0, N0, 0, zb0, nullptr, nullptr, T_b, N0, N0, 1.f, nullptr, 0);        // L0 @ z0
    gemm(T_b, 128, 0, Wdb, nullptr, out0, nullptr, N0, 128, 1.f, nullptr, 2);
    gemm(B1b, N1, 0, zb1, nullptr, nullptr, T_b, N0, N1, 1.f, nullptr, 0);        // B1 @ z1
    gemm(T_b, 128, 0, Wb1b, nullptr, out0, nullptr, N0, 128, 1.f, nullptr, 3);    // + relu

    // ================= z1_out =================
    attn_uv<<<dim3(N1 / 4), dim3(128), 0, stream>>>(z1, N1, adw, uvec, vvec);
    attn_mat<<<dim3(N1 / 256, N1), dim3(256), 0, stream>>>(L1d, N1, uvec, vvec, adb, bigb);
    gemm(bigb, N1, 0, zb1, nullptr, nullptr, T_b, N1, N1, 1.f, nullptr, 0);       // A_d @ z1
    gemm(T_b, 128, 0, Wdb, nullptr, out1, nullptr, N1, 128, 1.f, nullptr, 0);     // init

    attn_uv<<<dim3(N1 / 4), dim3(128), 0, stream>>>(z1, N1, auw, uvec, vvec);
    attn_mat<<<dim3(N1 / 256, N1), dim3(256), 0, stream>>>(L1u, N1, uvec, vvec, aub, bigb);
    gemm(bigb, N1, 0, zb1, nullptr, nullptr, T_b, N1, N1, 1.f, nullptr, 0);       // A_u @ z1
    gemm(T_b, 128, 0, Wub, nullptr, out1, nullptr, N1, 128, 1.f, nullptr, 2);

    gemm(B1b, N1, 1, zb0, nullptr, nullptr, T_b, N1, N0, 1.f, nullptr, 0);        // B1^T @ z0
    gemm(T_b, 128, 0, Wb1b, nullptr, out1, nullptr, N1, 128, 1.f, nullptr, 2);
    gemm(B2b, N2, 0, zb2, nullptr, nullptr, T_b, N1, N2, 1.f, nullptr, 0);        // B2 @ z2
    gemm(T_b, 128, 0, Wb2b, nullptr, out1, nullptr, N1, 128, 1.f, nullptr, 2);

    addmat_bf16<<<dim3(2048), dim3(256), 0, stream>>>(L1d, L1u, bigb, (size_t)N1 * N1);
    power_eps(bigb, N1);
    const unsigned short* yf1 = harmonic(bigb, N1, z1, zb1);
    gemm(yf1, 128, 0, Whb, nullptr, out1, nullptr, N1, 128, 1.f, nullptr, 3);     // + relu

    // ================= z2_out =================
    power_eps(Lb2, N2);
    const unsigned short* yf2 = harmonic(Lb2, N2, z2, zb2);
    gemm(yf2, 128, 0, Whb, nullptr, out2, nullptr, N2, 128, 1.f, nullptr, 0);     // init
    gemm(Lb2, N2, 0, zb2, nullptr, nullptr, T_b, N2, N2, 1.f, nullptr, 0);        // L2 @ z2
    gemm(T_b, 128, 0, Wub, nullptr, out2, nullptr, N2, 128, 1.f, nullptr, 2);
    gemm(B2b, N2, 1, zb1, nullptr, nullptr, T_b, N2, N1, 1.f, nullptr, 0);        // B2^T @ z1
    gemm(T_b, 128, 0, Wb2b, nullptr, out2, nullptr, N2, 128, 1.f, nullptr, 3);    // + relu
}
